// MultiHeadedAttention_26079041421457
// MI455X (gfx1250) — compile-verified, matrix operand order checked by reading, not run on hardware
//
#include <hip/hip_runtime.h>

// ---------------------------------------------------------------------------
// MultiHeadedAttention for MI455X (gfx1250, wave32, WMMA bf16)
// B=4, S=2048, D=1024, H=16, DK=64
// Pipeline:
//   gemm_bias<f32in,bf16out> x3  -> Q,K,V (bf16, [B,S,D], head h at cols h*64..)
//   attn (flash-style, bf16 WMMA, f32 online softmax, TDM K-tile staging)
//   gemm_bias<bf16in,f32out>     -> d_out = attn_out @ Wo^T + bo
// ---------------------------------------------------------------------------

typedef __attribute__((ext_vector_type(16))) __bf16 v16bf;
typedef __attribute__((ext_vector_type(2)))  __bf16 v2bf;
typedef __attribute__((ext_vector_type(8)))  float  v8f;
typedef __attribute__((ext_vector_type(4)))  unsigned int v4u;
typedef __attribute__((ext_vector_type(8)))  int    v8i;
typedef __attribute__((ext_vector_type(4)))  int    v4i;
using u16 = unsigned short;
using u32 = unsigned int;
using u64 = unsigned long long;

struct __align__(16) Q4 { u32 x, y, z, w; };
struct __align__(16) F4 { float x, y, z, w; };
struct Q8 { Q4 a, b; };

__device__ __forceinline__ u16 f2bf(float f) {
  u32 u = __builtin_bit_cast(u32, f);
  u += 0x7FFFu + ((u >> 16) & 1u);   // round-to-nearest-even
  return (u16)(u >> 16);
}
__device__ __forceinline__ u32 pack2(float a, float b) {
#if __has_builtin(__builtin_amdgcn_cvt_pk_bf16_f32)
  return __builtin_bit_cast(u32, __builtin_amdgcn_cvt_pk_bf16_f32(a, b));
#else
  return (u32)f2bf(a) | ((u32)f2bf(b) << 16);
#endif
}

// Load one 16x32 bf16 A/B fragment: element e<8 at k=kb+e, e>=8 at k=kb+16+(e-8).
__device__ __forceinline__ v16bf load_frag(const u16* p0, const u16* p1) {
  Q8 r;
  r.a = *(const Q4*)p0;
  r.b = *(const Q4*)p1;
  return __builtin_bit_cast(v16bf, r);
}

__device__ __forceinline__ v8f wmma_bf16(v16bf a, v16bf b, v8f c) {
  return __builtin_amdgcn_wmma_f32_16x16x32_bf16(
      false, a, false, b, (short)0, c, false, false);
}

// ---------------------------------------------------------------------------
// Tensor Data Mover: 2D bf16 tile load Global->LDS with LDS row padding.
// tile: tile_d0 elems/row (row = dim0), tile_d1 rows; global row stride
// stride_elems; LDS padded by pad 16B after every 128B (matches [..][72] u16).
// ---------------------------------------------------------------------------
#if __has_builtin(__builtin_amdgcn_tensor_load_to_lds) && \
    __has_builtin(__builtin_amdgcn_s_wait_tensorcnt)
#define HAVE_TDM 1
__device__ __forceinline__ void tdm_load_tile_64x64_bf16(
    u32 lds_addr, const u16* gaddr, u32 stride_elems) {
  const u64 ga = (u64)(size_t)gaddr;
  v4u g0;
  g0[0] = 1u;                                      // count=1, user mode, no gather
  g0[1] = lds_addr;                                // lds_addr [63:32]
  g0[2] = (u32)ga;                                 // global_addr[31:0]
  g0[3] = (u32)((ga >> 32) & 0x1FFFFFFu) | (2u << 30);  // addr[56:32], type=2
  v8i g1;
  // mask=0 | data_size=1(2B)<<16 | pad_en<<20 | pad_interval(32dw)=4<<22 |
  // pad_amount(4dw)=3<<25
  g1[0] = (int)((1u << 16) | (1u << 20) | (4u << 22) | (3u << 25));
  g1[1] = 0;                                       // abar=0, tensor_dim0[15:0]=0
  g1[2] = (int)0x4000u;                            // tensor_dim0=1<<30, dim1 lo=0
  g1[3] = (int)(0x4000u | (64u << 16));            // tensor_dim1=1<<30, tile_dim0=64
  g1[4] = 64;                                      // tile_dim1=64, tile_dim2=0
  g1[5] = (int)stride_elems;                       // tensor_dim0_stride[31:0]
  g1[6] = 0;                                       // stride0 hi, stride1 lo
  g1[7] = 0;                                       // stride1 hi (unused, 2D)
#if __clang_major__ >= 23
  __builtin_amdgcn_tensor_load_to_lds(g0, g1, (v4i)0, (v4i)0, (v8i)0, 0);
#else
  __builtin_amdgcn_tensor_load_to_lds(g0, g1, (v4i)0, (v4i)0, 0);
#endif
}
#endif

// ---------------------------------------------------------------------------
// GEMM: Out[m,n] = sum_k A[m,k] * W[n,k] + bias[n]   (K = N = 1024)
// Block: 256 threads, tile 128x128, BK=64. 8 waves in 2x4; wave tile 64x32.
// ---------------------------------------------------------------------------
template <bool A_F32, bool OUT_BF16>
__global__ __launch_bounds__(256) void gemm_bias(
    const void* __restrict__ Av, const float* __restrict__ W,
    const float* __restrict__ bias, void* __restrict__ Outv, int M) {
  constexpr int KD = 1024, ND = 1024;
  __shared__ u16 As[128][72];   // 144B row stride: frag ds_load_b128 conflict-free
  __shared__ u16 Bs[128][72];

  const int n0 = blockIdx.x * 128;
  const int m0 = blockIdx.y * 128;
  const int t  = threadIdx.x;
  const int w  = t >> 5, l = t & 31;
  const int wm = w >> 2, wn = w & 3;      // wave grid 2 x 4
  const int lr = l & 15, lh = l >> 4;
  const int kb0 = lh << 3;                // 0 or 8

  v8f acc[4][2] = {};

  for (int k0 = 0; k0 < KD; k0 += 64) {
    __syncthreads();
    // ---- stage 128x64 A tile and 128x64 B tile (convert f32->bf16 inline) ---
#pragma unroll
    for (int it = 0; it < 4; ++it) {
      const int c   = t + it * 256;       // 8-element chunk id
      const int row = c >> 3;             // 0..127
      const int c8  = (c & 7) << 3;       // 0..56
      if (A_F32) {
        const float* ap = (const float*)Av + (size_t)(m0 + row) * KD + k0 + c8;
        if (k0 + 64 < KD) __builtin_prefetch(ap + 64, 0, 3);  // global_prefetch_b8
        F4 f0 = *(const F4*)ap;
        F4 f1 = *(const F4*)(ap + 4);
        Q4 pk;
        pk.x = pack2(f0.x, f0.y); pk.y = pack2(f0.z, f0.w);
        pk.z = pack2(f1.x, f1.y); pk.w = pack2(f1.z, f1.w);
        *(Q4*)&As[row][c8] = pk;
      } else {
        const u16* ap = (const u16*)Av + (size_t)(m0 + row) * KD + k0 + c8;
        if (k0 + 64 < KD) __builtin_prefetch(ap + 64, 0, 3);
        *(Q4*)&As[row][c8] = *(const Q4*)ap;
      }
      const float* wp = W + (size_t)(n0 + row) * KD + k0 + c8;
      if (k0 + 64 < KD) __builtin_prefetch(wp + 64, 0, 3);
      F4 g0 = *(const F4*)wp;
      F4 g1 = *(const F4*)(wp + 4);
      Q4 pb;
      pb.x = pack2(g0.x, g0.y); pb.y = pack2(g0.z, g0.w);
      pb.z = pack2(g1.x, g1.y); pb.w = pack2(g1.z, g1.w);
      *(Q4*)&Bs[row][c8] = pb;
    }
    __syncthreads();
    // ---- 2 k-steps of 32, 8 WMMAs each ------------------------------------
#pragma unroll
    for (int ks = 0; ks < 64; ks += 32) {
      v16bf af[4], bf[2];
#pragma unroll
      for (int i = 0; i < 4; ++i) {
        const u16* r = &As[wm * 64 + i * 16 + lr][ks + kb0];
        af[i] = load_frag(r, r + 16);
      }
#pragma unroll
      for (int j = 0; j < 2; ++j) {
        const u16* r = &Bs[wn * 32 + j * 16 + lr][ks + kb0];
        bf[j] = load_frag(r, r + 16);
      }
#pragma unroll
      for (int i = 0; i < 4; ++i)
#pragma unroll
        for (int j = 0; j < 2; ++j)
          acc[i][j] = wmma_bf16(af[i], bf[j], acc[i][j]);
    }
  }

  // ---- epilogue: + bias, store --------------------------------------------
#pragma unroll
  for (int j = 0; j < 2; ++j) {
    const int ncol = n0 + wn * 32 + j * 16 + lr;
    const float bv = bias[ncol];
#pragma unroll
    for (int i = 0; i < 4; ++i) {
      const int mrow = m0 + wm * 64 + i * 16 + lh * 8;
#pragma unroll
      for (int v = 0; v < 8; ++v) {
        const float val = acc[i][j][v] + bv;
        if (OUT_BF16)
          ((u16*)Outv)[(size_t)(mrow + v) * ND + ncol] = f2bf(val);
        else
          ((float*)Outv)[(size_t)(mrow + v) * ND + ncol] = val;
      }
    }
  }
}

// ---------------------------------------------------------------------------
// Flash attention. Grid: B*H*(S/128) workgroups of 256 threads (8 waves).
// Wave owns 16 q-rows; streams 64-key K/V tiles; online softmax in f32.
// K tile staged by the TDM (wave 0) while waves 1..7 transpose-stage V.
// ---------------------------------------------------------------------------
__global__ __launch_bounds__(256) void attn(
    const u16* __restrict__ Q, const u16* __restrict__ K,
    const u16* __restrict__ V, u16* __restrict__ O) {
  constexpr int S = 2048, D = 1024, DK = 64;
  __shared__ u16 Ks[64][72];       // [key][dk]
  __shared__ u16 Vs[64][72];       // [dk][key]  (transposed for PV B-frags)
  __shared__ u16 Ps[8][16][72];    // per-wave P tile [qrow][key] in bf16

  const int wg = blockIdx.x;
  const int qt = wg & 15;          // S/128 = 16 q-tiles
  const int bh = wg >> 4;
  const int b  = bh >> 4;          // H = 16
  const int h  = bh & 15;

  const size_t baseBH = (size_t)b * S * D + (size_t)h * DK;
  const u16* Qb = Q + baseBH;
  const u16* Kb = K + baseBH;
  const u16* Vb = V + baseBH;

  const int t = threadIdx.x;
  const int w = t >> 5, l = t & 31;
  const int lr = l & 15, lh = l >> 4;
  const int kb0 = lh << 3;

  // Q fragments for this wave's 16 rows (dk 0..31 and 32..63)
  const int qrow = qt * 128 + w * 16 + lr;
  v16bf qf[2];
#pragma unroll
  for (int ks = 0; ks < 2; ++ks) {
    const u16* p = Qb + (size_t)qrow * D + ks * 32 + kb0;
    qf[ks] = load_frag(p, p + 16);
  }

  v8f o[4] = {};
  float mrun[8], lrun[8];
#pragma unroll
  for (int v = 0; v < 8; ++v) { mrun[v] = -3.0e38f; lrun[v] = 0.f; }
  const float cl2 = 0.125f * 1.44269504f;   // 1/sqrt(DK) * log2(e)

  for (int kv = 0; kv < S / 64; ++kv) {
    const int key0 = kv * 64;
    __syncthreads();
    // ---- stage K [key][dk] (TDM if available) and V^T [dk][key] ------------
#ifdef HAVE_TDM
    if (w == 0) {
      tdm_load_tile_64x64_bf16((u32)(size_t)&Ks[0][0],
                               Kb + (size_t)key0 * D, D);
      __builtin_amdgcn_s_wait_tensorcnt(0);
    } else {
#pragma unroll
      for (int it = 0; it < 3; ++it) {
        const int c = (t - 32) + it * 224;     // 224 threads, 512 chunks
        if (c < 512) {
          const int row = c >> 3;              // key 0..63
          const int c8  = (c & 7) << 3;        // dk base
          Q4 vv = *(const Q4*)(Vb + (size_t)(key0 + row) * D + c8);
          u16 ve[8];
          ve[0] = (u16)vv.x; ve[1] = (u16)(vv.x >> 16);
          ve[2] = (u16)vv.y; ve[3] = (u16)(vv.y >> 16);
          ve[4] = (u16)vv.z; ve[5] = (u16)(vv.z >> 16);
          ve[6] = (u16)vv.w; ve[7] = (u16)(vv.w >> 16);
#pragma unroll
          for (int e = 0; e < 8; ++e) Vs[c8 + e][row] = ve[e];
        }
      }
    }
#else
#pragma unroll
    for (int it = 0; it < 2; ++it) {
      const int c   = t + it * 256;
      const int row = c >> 3;            // key 0..63
      const int c8  = (c & 7) << 3;      // dk base
      *(Q4*)&Ks[row][c8] = *(const Q4*)(Kb + (size_t)(key0 + row) * D + c8);
      Q4 vv = *(const Q4*)(Vb + (size_t)(key0 + row) * D + c8);
      u16 ve[8];
      ve[0] = (u16)vv.x; ve[1] = (u16)(vv.x >> 16);
      ve[2] = (u16)vv.y; ve[3] = (u16)(vv.y >> 16);
      ve[4] = (u16)vv.z; ve[5] = (u16)(vv.z >> 16);
      ve[6] = (u16)vv.w; ve[7] = (u16)(vv.w >> 16);
#pragma unroll
      for (int e = 0; e < 8; ++e) Vs[c8 + e][row] = ve[e];
    }
#endif
    // prefetch next KV tile while this one is consumed
    if (kv + 1 < S / 64) {
      const int row = t >> 2, c8 = (t & 3) << 4;
      __builtin_prefetch(Kb + (size_t)(key0 + 64 + row) * D + c8, 0, 3);
      __builtin_prefetch(Vb + (size_t)(key0 + 64 + row) * D + c8, 0, 3);
    }
    __syncthreads();

    // ---- scores: 4 frags of 16 keys, 2 WMMAs each --------------------------
    v8f sf[4];
#pragma unroll
    for (int j = 0; j < 4; ++j) {
      v8f a = {};
#pragma unroll
      for (int ks = 0; ks < 2; ++ks) {
        const u16* p = &Ks[j * 16 + lr][ks * 32 + kb0];
        a = wmma_bf16(qf[ks], load_frag(p, p + 16), a);
      }
      sf[j] = a;
    }

    // ---- online softmax ----------------------------------------------------
    float mnew[8], alpha[8], psum[8];
#pragma unroll
    for (int v = 0; v < 8; ++v) {
      float mx = fmaxf(fmaxf(sf[0][v], sf[1][v]), fmaxf(sf[2][v], sf[3][v]));
#pragma unroll
      for (int msk = 1; msk < 16; msk <<= 1)
        mx = fmaxf(mx, __shfl_xor(mx, msk, 32));
      mnew[v]  = fmaxf(mrun[v], mx);
      alpha[v] = exp2f(cl2 * (mrun[v] - mnew[v]));
      psum[v]  = 0.f;
    }
#pragma unroll
    for (int j = 0; j < 4; ++j)
#pragma unroll
      for (int v = 0; v < 8; ++v) {
        const float p = exp2f(cl2 * (sf[j][v] - mnew[v]));
        psum[v] += p;
        Ps[w][lh * 8 + v][j * 16 + lr] = f2bf(p);
      }
#pragma unroll
    for (int v = 0; v < 8; ++v) {
      float s = psum[v];
#pragma unroll
      for (int msk = 1; msk < 16; msk <<= 1) s += __shfl_xor(s, msk, 32);
      lrun[v] = lrun[v] * alpha[v] + s;
      mrun[v] = mnew[v];
    }
#pragma unroll
    for (int n = 0; n < 4; ++n)
#pragma unroll
      for (int v = 0; v < 8; ++v) o[n][v] *= alpha[v];

    // ---- PV: O += P @ V ----------------------------------------------------
#pragma unroll
    for (int ks = 0; ks < 2; ++ks) {
      const u16* pp = &Ps[w][lr][ks * 32 + kb0];
      const v16bf pf = load_frag(pp, pp + 16);
#pragma unroll
      for (int n = 0; n < 4; ++n) {
        const u16* vp = &Vs[n * 16 + lr][ks * 32 + kb0];
        o[n] = wmma_bf16(pf, load_frag(vp, vp + 16), o[n]);
      }
    }
  }

  // ---- normalize and store bf16 attn output [B,S,D] ------------------------
#pragma unroll
  for (int n = 0; n < 4; ++n)
#pragma unroll
    for (int v = 0; v < 8; ++v) {
      const float val = o[n][v] / lrun[v];
      const size_t row = (size_t)b * S + qt * 128 + w * 16 + lh * 8 + v;
      O[row * D + h * DK + n * 16 + lr] = f2bf(val);
    }
}

// ---------------------------------------------------------------------------
extern "C" void kernel_launch(void* const* d_in, const int* in_sizes, int n_in,
                              void* d_out, int out_size, void* d_ws, size_t ws_size,
                              hipStream_t stream) {
  constexpr size_t MD = (size_t)8192 * 1024;  // B*S x D elements
  const float* query = (const float*)d_in[0];
  const float* key   = (const float*)d_in[1];
  const float* value = (const float*)d_in[2];
  const float* Wq = (const float*)d_in[3];  const float* bq = (const float*)d_in[4];
  const float* Wk = (const float*)d_in[5];  const float* bk = (const float*)d_in[6];
  const float* Wv = (const float*)d_in[7];  const float* bv = (const float*)d_in[8];
  const float* Wo = (const float*)d_in[9];  const float* bo = (const float*)d_in[10];

  u16* qbf  = (u16*)d_ws;          // bf16 workspace: 4 x 16.8 MB = 67 MB
  u16* kbf  = qbf + MD;
  u16* vbf  = kbf + MD;
  u16* aout = vbf + MD;

  dim3 blk(256);
  dim3 gg(1024 / 128, 8192 / 128);   // (N tiles, M tiles)

  gemm_bias<true, true><<<gg, blk, 0, stream>>>(query, Wq, bq, qbf, 8192);
  gemm_bias<true, true><<<gg, blk, 0, stream>>>(key,   Wk, bk, kbf, 8192);
  gemm_bias<true, true><<<gg, blk, 0, stream>>>(value, Wv, bv, vbf, 8192);

  attn<<<dim3(4 * 16 * 16), blk, 0, stream>>>(qbf, kbf, vbf, aout);

  gemm_bias<false, false><<<gg, blk, 0, stream>>>(aout, Wo, bo, (float*)d_out, 8192);
}
